// SftLayer_15530601742455
// MI455X (gfx1250) — matrix operand order repaired and checked by reading, not compiled, not run on hardware
//
#include <hip/hip_runtime.h>
#include <hip/hip_bf16.h>

// -------- types --------
typedef __bf16 bf16;
typedef __attribute__((ext_vector_type(16))) __bf16 v16bf;
typedef __attribute__((ext_vector_type(8)))  __bf16 v8bf;
typedef __attribute__((ext_vector_type(4)))  __bf16 v4bf;
typedef __attribute__((ext_vector_type(8)))  float  v8f;

#define DEVFN static __device__ __forceinline__

constexpr int Bc = 16, Nc = 128, Dc = 128, Ec = 128, Fc = 2048, Hc = 8, DHc = 16;
constexpr int LDA = 136;                 // LDS row stride in bf16 (mult of 8 -> 16B aligned chunks)

// ---------------- WMMA fragment helpers ----------------
// A fragment (16x32 bf16, M = lane&15):
//   lanes 0-15 : K = {0..7, 16..23}; lanes 16-31 : K = {8..15, 24..31}
DEVFN v16bf ldsAFrag(const bf16* s, int rowBase, int k0, int lane) {
  const int m = lane & 15, h = lane >> 4;
  const v8bf* p = (const v8bf*)(s + (rowBase + m) * LDA + k0 + h * 8);
  v8bf lo = p[0];           // K in [k0+8h, k0+8h+8)
  v8bf hi = p[2];           // K in [k0+16+8h, ...)
  v16bf a;
#pragma unroll
  for (int e = 0; e < 8; ++e) { a[e] = lo[e]; a[8 + e] = hi[e]; }
  return a;
}

// B fragment (32x16 bf16): lane n holds column n; lanes 0-15 K = {0..7, 16..23}; lanes 16-31 : K = {8..15, 24..31}
// B[k][n] = W[nBase+n][k]  (weight stored row-major [out][in])
DEVFN v16bf gblBFrag(const bf16* __restrict__ W, int ldw, int nBase, int k0, int lane) {
  const int n = lane & 15, h = lane >> 4;
  const v8bf* p = (const v8bf*)(W + (size_t)(nBase + n) * ldw + k0 + h * 8);
  v8bf lo = p[0], hi = p[2];
  v16bf b;
#pragma unroll
  for (int e = 0; e < 8; ++e) { b[e] = lo[e]; b[8 + e] = hi[e]; }
  return b;
}

DEVFN v8f wmma_bf16(v16bf a, v16bf b, v8f c) {
  return __builtin_amdgcn_wmma_f32_16x16x32_bf16(false, a, false, b, (short)0, c, false, false);
}

// wave computes its 16 rows x 128 cols, K=128 starting at weight column kcol0
DEVFN void gemmRowBlock(const v16bf aF[4], const bf16* __restrict__ W, int ldw, int kcol0,
                        v8f acc[8], bool accum, int lane) {
#pragma unroll
  for (int t = 0; t < 8; ++t) {
    v8f c = {0.f, 0.f, 0.f, 0.f, 0.f, 0.f, 0.f, 0.f};
    if (accum) c = acc[t];
#pragma unroll
    for (int kb = 0; kb < 4; ++kb) {
      v16bf bF = gblBFrag(W, ldw, 16 * t, kcol0 + 32 * kb, lane);
      c = wmma_bf16(aF[kb], bF, c);
    }
    acc[t] = c;
  }
}

// ---------------- LDS staging: 128x128 f32 tile -> bf16 LDS (float4 loads) ----------------
DEVFN void fillTileBf16(const float* __restrict__ src, bf16* dst, int tid) {
  const float4* s4 = (const float4*)src;
  for (int idx = tid; idx < 128 * 32; idx += 256) {
    const int r = idx >> 5, c4 = idx & 31;
    float4 f = s4[idx];
    v4bf o;
    o[0] = (bf16)f.x; o[1] = (bf16)f.y; o[2] = (bf16)f.z; o[3] = (bf16)f.w;
    *(v4bf*)(dst + r * LDA + c4 * 4) = o;
  }
}

// ---------------- reductions / LayerNorm ----------------
DEVFN float halfSum(float v) {      // reduce over 16 lanes within each wave half
  v += __shfl_xor(v, 1, 32);
  v += __shfl_xor(v, 2, 32);
  v += __shfl_xor(v, 4, 32);
  v += __shfl_xor(v, 8, 32);
  return v;
}
DEVFN float waveMax(float v) {
#pragma unroll
  for (int m = 1; m < 32; m <<= 1) v = fmaxf(v, __shfl_xor(v, m, 32));
  return v;
}
DEVFN float waveSum(float v) {
#pragma unroll
  for (int m = 1; m < 32; m <<= 1) v += __shfl_xor(v, m, 32);
  return v;
}

// vals[t] = element at column t*16+colLo of one full 128-wide row (row shared by a wave half)
DEVFN void rowLN(float vals[8], const float* __restrict__ g, const float* __restrict__ bta,
                 int colLo) {
  float s = 0.f;
#pragma unroll
  for (int t = 0; t < 8; ++t) s += vals[t];
  s = halfSum(s);
  const float mu = s * (1.0f / 128.0f);
  float v2 = 0.f;
#pragma unroll
  for (int t = 0; t < 8; ++t) { float d = vals[t] - mu; v2 += d * d; }
  v2 = halfSum(v2);
  const float rs = rsqrtf(v2 * (1.0f / 128.0f) + 1e-5f);
#pragma unroll
  for (int t = 0; t < 8; ++t) {
    int col = t * 16 + colLo;
    vals[t] = (vals[t] - mu) * rs * g[col] + bta[col];
  }
}

// ---------------- weight conversion ----------------
__global__ void cvt_kernel(const float* __restrict__ src, bf16* __restrict__ dst, int n) {
  for (int i = blockIdx.x * blockDim.x + threadIdx.x; i < n; i += gridDim.x * blockDim.x)
    dst[i] = (bf16)src[i];
}
__global__ void cvt_wm_kernel(const float* __restrict__ Wm, bf16* __restrict__ dE,
                              bf16* __restrict__ dS, bf16* __restrict__ dT) {
  int i = blockIdx.x * blockDim.x + threadIdx.x;
  if (i < 128 * 384) {
    int r = i / 384, c = i % 384;
    bf16 v = (bf16)Wm[i];
    if (c < 128)      dE[r * 128 + c] = v;
    else if (c < 256) dS[r * 128 + (c - 128)] = v;
    else              dT[r * 128 + (c - 256)] = v;
  }
}

// ---------------- kernel A: ns / nt / q projections ----------------
__global__ __launch_bounds__(256) void node_proj_kernel(
    const float* __restrict__ node,
    const bf16* __restrict__ WmS, const bf16* __restrict__ WmT, const bf16* __restrict__ WqB,
    const float* __restrict__ bq,
    float* __restrict__ nsv, float* __restrict__ ntv, float* __restrict__ qv) {
  __shared__ bf16 sA[128 * LDA];
  const int b = blockIdx.x;
  fillTileBf16(node + (size_t)b * Nc * Dc, sA, threadIdx.x);
  __syncthreads();

  const int lane = threadIdx.x & 31, w = threadIdx.x >> 5;
  const int colLo = lane & 15, hiHalf = lane >> 4;
  v16bf aF[4];
#pragma unroll
  for (int kb = 0; kb < 4; ++kb) aF[kb] = ldsAFrag(sA, 16 * w, 32 * kb, lane);

  v8f acc[8];
  // ns = node @ WmS^T  (straight-line: keeps weight pointers in global addrspace)
  gemmRowBlock(aF, WmS, Dc, 0, acc, false, lane);
#pragma unroll
  for (int r = 0; r < 8; ++r) {
    const int jj = 16 * w + hiHalf * 8 + r;
#pragma unroll
    for (int t = 0; t < 8; ++t)
      nsv[((size_t)b * Nc + jj) * Dc + t * 16 + colLo] = acc[t][r];
  }
  // nt = node @ WmT^T
  gemmRowBlock(aF, WmT, Dc, 0, acc, false, lane);
#pragma unroll
  for (int r = 0; r < 8; ++r) {
    const int jj = 16 * w + hiHalf * 8 + r;
#pragma unroll
    for (int t = 0; t < 8; ++t)
      ntv[((size_t)b * Nc + jj) * Dc + t * 16 + colLo] = acc[t][r];
  }
  // q = node @ Wq^T + bq
  gemmRowBlock(aF, WqB, Dc, 0, acc, false, lane);
#pragma unroll
  for (int r = 0; r < 8; ++r) {
    const int jj = 16 * w + hiHalf * 8 + r;
#pragma unroll
    for (int t = 0; t < 8; ++t) {
      int col = t * 16 + colLo;
      qv[((size_t)b * Nc + jj) * Dc + col] = acc[t][r] + bq[col];
    }
  }
}

// ---------------- kernel B: fused edge pipeline + attention ----------------
// one workgroup per (b,i): GEMM pre -> LN/ReLU(memory) -> GEMM pe -> LN/ReLU -> +edge -> LN
// -> edge_new; GEMM k -> scores/softmax; GEMM v -> ctx
__global__ __launch_bounds__(256) void edge_kernel(
    const float* __restrict__ edge, const float* __restrict__ nsv,
    const float* __restrict__ ntv, const float* __restrict__ qv,
    const unsigned char* __restrict__ node_mask,
    const bf16* __restrict__ WmE, const bf16* __restrict__ WeB,
    const bf16* __restrict__ WkB, const bf16* __restrict__ WvB,
    const float* __restrict__ bm, const float* __restrict__ gm, const float* __restrict__ btm,
    const float* __restrict__ be, const float* __restrict__ gpe, const float* __restrict__ bpe,
    const float* __restrict__ gne, const float* __restrict__ bne,
    const float* __restrict__ bk, const float* __restrict__ bv,
    float* __restrict__ out_edge, float* __restrict__ ctx_out) {
  __shared__ bf16 buf0[128 * LDA];   // edge(bf16) -> k -> v
  __shared__ bf16 buf1[128 * LDA];   // memory(bf16)
  __shared__ float sQ[128];
  __shared__ float sNS[128];
  __shared__ float sScore[Hc][Nc];

  const int bi = blockIdx.x;           // b*N + i
  const int b = bi >> 7;
  const size_t edgeBase = (size_t)bi * (Nc * Ec);
  const int tid = threadIdx.x;
  const int lane = tid & 31, w = tid >> 5;
  const int colLo = lane & 15, hiHalf = lane >> 4;

  fillTileBf16(edge + edgeBase, buf0, tid);
  if (tid < 128) sQ[tid] = qv[(size_t)bi * Dc + tid];
  else           sNS[tid - 128] = nsv[(size_t)bi * Dc + (tid - 128)];
  __syncthreads();

  v16bf aF[4];
#pragma unroll
  for (int kb = 0; kb < 4; ++kb) aF[kb] = ldsAFrag(buf0, 16 * w, 32 * kb, lane);
  v8f acc[8];

  // ---- GEMM1: pre = edge @ WmE^T ; epilogue: + ns + nt + bm, LN, ReLU -> memory ----
  gemmRowBlock(aF, WmE, Ec, 0, acc, false, lane);
#pragma unroll
  for (int r = 0; r < 8; ++r) {
    const int jj = 16 * w + hiHalf * 8 + r;
    float vals[8];
#pragma unroll
    for (int t = 0; t < 8; ++t) {
      int col = t * 16 + colLo;
      vals[t] = acc[t][r] + sNS[col] + ntv[((size_t)b * Nc + jj) * Dc + col] + bm[col];
    }
    rowLN(vals, gm, btm, colLo);
#pragma unroll
    for (int t = 0; t < 8; ++t)
      buf1[jj * LDA + t * 16 + colLo] = (bf16)fmaxf(vals[t], 0.f);
  }
  // memory rows are wave-local (LDS per-wave in-order) -> reload A frags directly
#pragma unroll
  for (int kb = 0; kb < 4; ++kb) aF[kb] = ldsAFrag(buf1, 16 * w, 32 * kb, lane);

  // ---- GEMM2: pe = memory @ We^T ; LN,ReLU, + edge, LN -> edge_new ----
  gemmRowBlock(aF, WeB, Dc, 0, acc, false, lane);
#pragma unroll
  for (int r = 0; r < 8; ++r) {
    const int jj = 16 * w + hiHalf * 8 + r;
    float vals[8];
#pragma unroll
    for (int t = 0; t < 8; ++t) vals[t] = acc[t][r] + be[t * 16 + colLo];
    rowLN(vals, gpe, bpe, colLo);
    float v2[8];
#pragma unroll
    for (int t = 0; t < 8; ++t) {
      int col = t * 16 + colLo;
      v2[t] = fmaxf(vals[t], 0.f) + edge[edgeBase + (size_t)jj * Ec + col];
    }
    rowLN(v2, gne, bne, colLo);
#pragma unroll
    for (int t = 0; t < 8; ++t)
      out_edge[edgeBase + (size_t)jj * Ec + t * 16 + colLo] = v2[t];
  }

  // ---- GEMM3: k = memory @ Wk^T + bk -> buf0 ----
  gemmRowBlock(aF, WkB, Dc, 0, acc, false, lane);
#pragma unroll
  for (int r = 0; r < 8; ++r) {
    const int jj = 16 * w + hiHalf * 8 + r;
#pragma unroll
    for (int t = 0; t < 8; ++t) {
      int col = t * 16 + colLo;
      buf0[jj * LDA + col] = (bf16)(acc[t][r] + bk[col]);
    }
  }
  __syncthreads();

  // ---- scores = q.k / sqrt(DH), mask ----
  for (int idx = tid; idx < Hc * Nc; idx += 256) {
    int h = idx >> 7, j = idx & 127;
    float s = 0.f;
#pragma unroll
    for (int d = 0; d < DHc; ++d) s += sQ[h * DHc + d] * (float)buf0[j * LDA + h * DHc + d];
    s *= 0.25f;                                   // 1/sqrt(16)
    s = node_mask[b * Nc + j] ? s : -1.0e9f;
    sScore[h][j] = s;
  }
  __syncthreads();

  // ---- softmax over j: wave w owns head w ----
  {
    float m = -3.0e38f;
#pragma unroll
    for (int k4 = 0; k4 < 4; ++k4) m = fmaxf(m, sScore[w][lane + 32 * k4]);
    m = waveMax(m);
    float ssum = 0.f;
    float ev[4];
#pragma unroll
    for (int k4 = 0; k4 < 4; ++k4) {
      float e = __expf(sScore[w][lane + 32 * k4] - m);
      ev[k4] = e; ssum += e;
    }
    ssum = waveSum(ssum);
    const float inv = 1.f / ssum;
#pragma unroll
    for (int k4 = 0; k4 < 4; ++k4) sScore[w][lane + 32 * k4] = ev[k4] * inv;
  }
  __syncthreads();

  // ---- GEMM4: v = memory @ Wv^T + bv -> buf0 ----
  gemmRowBlock(aF, WvB, Dc, 0, acc, false, lane);
#pragma unroll
  for (int r = 0; r < 8; ++r) {
    const int jj = 16 * w + hiHalf * 8 + r;
#pragma unroll
    for (int t = 0; t < 8; ++t) {
      int col = t * 16 + colLo;
      buf0[jj * LDA + col] = (bf16)(acc[t][r] + bv[col]);
    }
  }
  __syncthreads();

  // ---- ctx[d] = sum_j attn[h(d), j] * v[j, d] ----
  if (tid < Dc) {
    const int d = tid, h = d >> 4;
    float c = 0.f;
    for (int j = 0; j < Nc; ++j) c += sScore[h][j] * (float)buf0[j * LDA + d];
    ctx_out[(size_t)bi * Dc + d] = c;
  }
}

// ---------------- kernel C: Wo + LN, FFN, final LN ----------------
__global__ __launch_bounds__(256) void out_kernel(
    const float* __restrict__ ctxv, const float* __restrict__ node,
    const bf16* __restrict__ WoB, const bf16* __restrict__ W1B, const bf16* __restrict__ W2B,
    const float* __restrict__ bo, const float* __restrict__ g2, const float* __restrict__ bt2,
    const float* __restrict__ b1, const float* __restrict__ b2,
    const float* __restrict__ g3, const float* __restrict__ bt3,
    float* __restrict__ out_x) {
  __shared__ bf16 bufA[128 * LDA];   // ctx -> x (bf16)
  __shared__ bf16 bufB[128 * LDA];   // relu(ffn hidden chunk)
  const int b = blockIdx.x;
  fillTileBf16(ctxv + (size_t)b * Nc * Dc, bufA, threadIdx.x);
  __syncthreads();

  const int lane = threadIdx.x & 31, w = threadIdx.x >> 5;
  const int colLo = lane & 15, hiHalf = lane >> 4;
  v16bf aF[4];
#pragma unroll
  for (int kb = 0; kb < 4; ++kb) aF[kb] = ldsAFrag(bufA, 16 * w, 32 * kb, lane);

  // x = LN(node + ctx @ Wo^T + bo)
  v8f acc[8];
  gemmRowBlock(aF, WoB, Dc, 0, acc, false, lane);
#pragma unroll
  for (int r = 0; r < 8; ++r) {
    const int jj = 16 * w + hiHalf * 8 + r;
    float vals[8];
#pragma unroll
    for (int t = 0; t < 8; ++t) {
      int col = t * 16 + colLo;
      vals[t] = acc[t][r] + bo[col] + node[((size_t)b * Nc + jj) * Dc + col];
    }
    rowLN(vals, g2, bt2, colLo);
#pragma unroll
    for (int t = 0; t < 8; ++t)
      bufA[jj * LDA + t * 16 + colLo] = (bf16)vals[t];   // wave-local rows
  }
#pragma unroll
  for (int kb = 0; kb < 4; ++kb) aF[kb] = ldsAFrag(bufA, 16 * w, 32 * kb, lane);

  // ff = relu(x @ W1^T + b1) @ W2^T, accumulated over 16 F-chunks
  v8f acc2[8];
#pragma unroll
  for (int t = 0; t < 8; ++t) acc2[t] = (v8f){0.f, 0.f, 0.f, 0.f, 0.f, 0.f, 0.f, 0.f};
#pragma unroll 1
  for (int ch = 0; ch < Fc / 128; ++ch) {
    v8f acc1[8];
    gemmRowBlock(aF, W1B + (size_t)ch * 128 * Dc, Dc, 0, acc1, false, lane);
#pragma unroll
    for (int r = 0; r < 8; ++r) {
      const int jj = 16 * w + hiHalf * 8 + r;
#pragma unroll
      for (int t = 0; t < 8; ++t) {
        int col = t * 16 + colLo;
        bufB[jj * LDA + col] = (bf16)fmaxf(acc1[t][r] + b1[ch * 128 + col], 0.f);
      }
    }
    v16bf aH[4];
#pragma unroll
    for (int kb = 0; kb < 4; ++kb) aH[kb] = ldsAFrag(bufB, 16 * w, 32 * kb, lane);
    gemmRowBlock(aH, W2B, Fc, ch * 128, acc2, true, lane);
  }

  // x_out = LN(x + ff + b2)
#pragma unroll
  for (int r = 0; r < 8; ++r) {
    const int jj = 16 * w + hiHalf * 8 + r;
    float vals[8];
#pragma unroll
    for (int t = 0; t < 8; ++t) {
      int col = t * 16 + colLo;
      vals[t] = acc2[t][r] + b2[col] + (float)bufA[jj * LDA + col];
    }
    rowLN(vals, g3, bt3, colLo);
#pragma unroll
    for (int t = 0; t < 8; ++t)
      out_x[((size_t)b * Nc + jj) * Dc + t * 16 + colLo] = vals[t];
  }
}

// ---------------- host launcher ----------------
extern "C" void kernel_launch(void* const* d_in, const int* in_sizes, int n_in,
                              void* d_out, int out_size, void* d_ws, size_t ws_size,
                              hipStream_t stream) {
  const float* node = (const float*)d_in[0];
  const float* edge = (const float*)d_in[1];
  const unsigned char* node_mask = (const unsigned char*)d_in[2];
  // d_in[3] edge_mask: unused by the reference
  const float* Wm  = (const float*)d_in[4];
  const float* bm  = (const float*)d_in[5];
  const float* gm  = (const float*)d_in[6];
  const float* btm = (const float*)d_in[7];
  const float* We  = (const float*)d_in[8];
  const float* be  = (const float*)d_in[9];
  const float* gpe = (const float*)d_in[10];
  const float* bpe = (const float*)d_in[11];
  const float* gne = (const float*)d_in[12];
  const float* bne = (const float*)d_in[13];
  const float* Wq  = (const float*)d_in[14];
  const float* bq  = (const float*)d_in[15];
  const float* Wk  = (const float*)d_in[16];
  const float* bk  = (const float*)d_in[17];
  const float* Wv  = (const float*)d_in[18];
  const float* bv  = (const float*)d_in[19];
  const float* Wo  = (const float*)d_in[20];
  const float* bo  = (const float*)d_in[21];
  const float* W1  = (const float*)d_in[22];
  const float* b1  = (const float*)d_in[23];
  const float* W2  = (const float*)d_in[24];
  const float* b2  = (const float*)d_in[25];
  const float* g2  = (const float*)d_in[26];
  const float* bt2 = (const float*)d_in[27];
  const float* g3  = (const float*)d_in[28];
  const float* bt3 = (const float*)d_in[29];

  // workspace layout: bf16 weights, then f32 intermediates (~5.3 MB total)
  bf16* wsb = (bf16*)d_ws;
  bf16 *WmE_b = wsb + 0,      *WmS_b = wsb + 16384,  *WmT_b = wsb + 32768;
  bf16 *We_b  = wsb + 49152,  *Wk_b  = wsb + 65536,  *Wv_b  = wsb + 81920;
  bf16 *Wq_b  = wsb + 98304,  *Wo_b  = wsb + 114688;
  bf16 *W1_b  = wsb + 131072, *W2_b  = wsb + 393216;
  float* wsf  = (float*)(wsb + 655360);
  float *nsv = wsf, *ntv = wsf + 262144, *qv = wsf + 524288, *ctxv = wsf + 786432;

  float* out_x = (float*)d_out;
  float* out_edge = out_x + (size_t)Bc * Nc * Dc;

  cvt_wm_kernel<<<(128 * 384 + 255) / 256, 256, 0, stream>>>(Wm, WmE_b, WmS_b, WmT_b);
  cvt_kernel<<<64, 256, 0, stream>>>(We, We_b, 16384);
  cvt_kernel<<<64, 256, 0, stream>>>(Wk, Wk_b, 16384);
  cvt_kernel<<<64, 256, 0, stream>>>(Wv, Wv_b, 16384);
  cvt_kernel<<<64, 256, 0, stream>>>(Wq, Wq_b, 16384);
  cvt_kernel<<<64, 256, 0, stream>>>(Wo, Wo_b, 16384);
  cvt_kernel<<<512, 256, 0, stream>>>(W1, W1_b, 262144);
  cvt_kernel<<<512, 256, 0, stream>>>(W2, W2_b, 262144);

  node_proj_kernel<<<Bc, 256, 0, stream>>>(node, WmS_b, WmT_b, Wq_b, bq, nsv, ntv, qv);

  edge_kernel<<<Bc * Nc, 256, 0, stream>>>(edge, nsv, ntv, qv, node_mask,
      WmE_b, We_b, Wk_b, Wv_b, bm, gm, btm, be, gpe, bpe, gne, bne, bk, bv,
      out_edge, ctxv);

  out_kernel<<<Bc, 256, 0, stream>>>(ctxv, node, Wo_b, W1_b, W2_b,
      bo, g2, bt2, b1, b2, g3, bt3, out_x);
}
